// _ZarvanBlock_75273596830053
// MI455X (gfx1250) — compile-verified, matrix operand order checked by reading, not run on hardware
//
#include <hip/hip_runtime.h>
#include <hip/hip_bf16.h>
#include <math.h>

// ---------------------------------------------------------------------------
// Zarvan block on MI455X (gfx1250): all ten big GEMMs run on
// v_wmma_f32_16x16x32_bf16 (bf16 inputs, fp32 accumulate), tiles staged into
// LDS with CDNA5 async copies (global_load_async_to_lds_b128 + ASYNCcnt),
// intermediates kept in bf16 to halve HBM traffic.  Softmax/scan/LayerNorm
// reductions are fp32 VALU kernels (bandwidth-trivial vs ~293 GFLOP of GEMM).
// ---------------------------------------------------------------------------

typedef __attribute__((ext_vector_type(16))) __bf16 bf16x16;
typedef __attribute__((ext_vector_type(8)))  __bf16 bf16x8;
typedef __attribute__((ext_vector_type(4)))  __bf16 bf16x4;
typedef __attribute__((ext_vector_type(8)))  float  f32x8;
typedef __attribute__((ext_vector_type(4)))  float  f32x4;

#define PI_F 3.14159265358979323846f

enum { EPI_F32=0, EPI_BF16=1, EPI_GELU_F32=2, EPI_GELU_BF16=3, EPI_SIG_BF16=4, EPI_RESID_F32=5 };

__device__ __forceinline__ float gelu_f(float x){
  return 0.5f * x * (1.0f + erff(x * 0.70710678118654752f));   // exact (erf) gelu
}

// CDNA5 async global->LDS copy (tracked by ASYNCcnt); per-lane 16B copy.
__device__ __forceinline__ void async_b128(unsigned lds_addr, const void* gaddr){
  asm volatile("global_load_async_to_lds_b128 %0, %1, off"
               :: "v"(lds_addr), "v"(gaddr) : "memory");
}
__device__ __forceinline__ void wait_async0(){
  asm volatile("s_wait_asynccnt 0x0" ::: "memory");
}

// Load one 16x32 bf16 WMMA A/B fragment from an LDS tile (row-major, stride 40
// halves).  ISA layout: lanes 0-15 hold K=0..7 (vgpr0-3) and K=16..23 (vgpr4-7),
// lanes 16-31 hold K=8..15 and K=24..31 -> two ds_load_b128 per fragment.
__device__ __forceinline__ bf16x16 ld_frag(const __bf16* s, int row, int lane){
  const int ho = (lane >> 4) << 3;                 // 0 or 8 halves
  bf16x8 lo = *reinterpret_cast<const bf16x8*>(s + row*40 + ho);
  bf16x8 hi = *reinterpret_cast<const bf16x8*>(s + row*40 + 16 + ho);
  return __builtin_shufflevector(lo, hi, 0,1,2,3,4,5,6,7,8,9,10,11,12,13,14,15);
}

// ---------------------------------------------------------------------------
// WMMA GEMM:  out(M x N) = epi( A(M x K) @ WT^T + bias )   [WT is N x K bf16]
// Block tile 128x128, 8 waves (4 along M x 2 along N), each wave 32x64
// (2x4 WMMA frags).  Double-buffered LDS fed by async global->LDS copies.
// ---------------------------------------------------------------------------
template<typename AT, int EPI>
__global__ __launch_bounds__(256) void gemm_wmma(
    const AT* __restrict__ A, const __bf16* __restrict__ WT,
    const float* __restrict__ bias, const float* __restrict__ resid,
    void* __restrict__ outv, int M, int N, int K)
{
  __shared__ __align__(16) __bf16 sA[2][128*40];
  __shared__ __align__(16) __bf16 sB[2][128*40];
  const int t    = threadIdx.x;
  const int lane = t & 31;
  const int wv   = t >> 5;
  const int wm   = wv & 3;           // wave position along M (0..3, 32 rows each)
  const int wn   = wv >> 2;          // wave position along N (0..1, 64 cols each)
  const int m0   = blockIdx.x * 128;
  const int n0   = blockIdx.y * 128;
  const int nk   = K >> 5;
  constexpr bool A_IS_F32 = (sizeof(AT) == 4);

  f32x8 acc[2][4];
  #pragma unroll
  for (int i=0;i<2;i++)
    #pragma unroll
    for (int j=0;j<4;j++)
      #pragma unroll
      for (int q=0;q<8;q++) acc[i][j][q] = 0.f;

  // per-thread tile-copy descriptors: 128 rows x 4 chunks(8 halves) = 512
  // chunks -> 2 per thread.
  int rowc[2], colc[2];
  #pragma unroll
  for (int i=0;i<2;i++){ const int lin = t + i*256; rowc[i] = lin >> 2; colc[i] = (lin & 3)*8; }

  const __bf16* gB[2]; unsigned lB[2][2];
  #pragma unroll
  for (int i=0;i<2;i++){
    gB[i]    = WT + (size_t)(n0 + rowc[i])*K + colc[i];
    lB[0][i] = (unsigned)(uintptr_t)&sB[0][rowc[i]*40 + colc[i]];
    lB[1][i] = (unsigned)(uintptr_t)&sB[1][rowc[i]*40 + colc[i]];
  }
  const __bf16* gA[2]; unsigned lA[2][2];
  if constexpr (!A_IS_F32){
    #pragma unroll
    for (int i=0;i<2;i++){
      gA[i]    = reinterpret_cast<const __bf16*>(A) + (size_t)(m0 + rowc[i])*K + colc[i];
      lA[0][i] = (unsigned)(uintptr_t)&sA[0][rowc[i]*40 + colc[i]];
      lA[1][i] = (unsigned)(uintptr_t)&sA[1][rowc[i]*40 + colc[i]];
    }
  }

  f32x4 raF[4];                       // fp32 A staging (first GEMM only)
  auto ldgA_f32 = [&](int kk){
    #pragma unroll
    for (int i=0;i<4;i++){
      const int lin = t + i*256, row = lin >> 3, c = lin & 7;
      raF[i] = *reinterpret_cast<const f32x4*>(
          reinterpret_cast<const float*>(A) + (size_t)(m0+row)*K + kk + c*4);
    }
  };
  auto stsA_f32 = [&](int buf){
    #pragma unroll
    for (int i=0;i<4;i++){
      const int lin = t + i*256, row = lin >> 3, c = lin & 7;
      bf16x4 h;
      #pragma unroll
      for (int q=0;q<4;q++) h[q] = (__bf16)raF[i][q];
      *reinterpret_cast<bf16x4*>(&sA[buf][row*40 + c*4]) = h;
    }
  };
  auto issue_async = [&](int buf, int kk){
    #pragma unroll
    for (int i=0;i<2;i++) async_b128(lB[buf][i], gB[i] + kk);
    if constexpr (!A_IS_F32){
      #pragma unroll
      for (int i=0;i<2;i++) async_b128(lA[buf][i], gA[i] + kk);
    }
  };

  // prologue: fill buffer 0
  issue_async(0, 0);
  if constexpr (A_IS_F32){ ldgA_f32(0); stsA_f32(0); }
  wait_async0();
  __syncthreads();

  int cur = 0;
  for (int kt = 0; kt < nk; ++kt){
    const bool has = (kt+1 < nk);
    if (has){
      issue_async(cur^1, (kt+1) << 5);       // DMA next slab while computing
      if constexpr (A_IS_F32) ldgA_f32((kt+1) << 5);
    }
    bf16x16 af[2], bfr[4];
    #pragma unroll
    for (int fm=0; fm<2; fm++) af[fm]  = ld_frag(&sA[cur][0], wm*32 + fm*16 + (lane&15), lane);
    #pragma unroll
    for (int fn=0; fn<4; fn++) bfr[fn] = ld_frag(&sB[cur][0], wn*64 + fn*16 + (lane&15), lane);
    #pragma unroll
    for (int fm=0; fm<2; fm++)
      #pragma unroll
      for (int fn=0; fn<4; fn++)
        acc[fm][fn] = __builtin_amdgcn_wmma_f32_16x16x32_bf16(
            false, af[fm], false, bfr[fn], (short)0, acc[fm][fn], false, false);
    if (has){
      if constexpr (A_IS_F32) stsA_f32(cur^1);
      wait_async0();
      __syncthreads();
      cur ^= 1;
    }
  }

  // Epilogue: C/D layout -> lane (0-15|16-31) = N, vgpr q = M (0-7|8-15).
  #pragma unroll
  for (int fm=0; fm<2; fm++){
    #pragma unroll
    for (int fn=0; fn<4; fn++){
      const int n  = n0 + wn*64 + fn*16 + (lane & 15);
      const float bs = bias[n];
      #pragma unroll
      for (int q=0;q<8;q++){
        const int m = m0 + wm*32 + fm*16 + ((lane>>4)<<3) + q;
        const size_t idx = (size_t)m*N + n;
        float c = acc[fm][fn][q] + bs;
        if constexpr (EPI==EPI_F32)            reinterpret_cast<float*>(outv)[idx]  = c;
        else if constexpr (EPI==EPI_BF16)      reinterpret_cast<__bf16*>(outv)[idx] = (__bf16)c;
        else if constexpr (EPI==EPI_GELU_F32)  reinterpret_cast<float*>(outv)[idx]  = gelu_f(c);
        else if constexpr (EPI==EPI_GELU_BF16) reinterpret_cast<__bf16*>(outv)[idx] = (__bf16)gelu_f(c);
        else if constexpr (EPI==EPI_SIG_BF16)  reinterpret_cast<__bf16*>(outv)[idx] = (__bf16)(1.f/(1.f+__expf(-c)));
        else /* EPI_RESID_F32 */               reinterpret_cast<float*>(outv)[idx]  = c + resid[idx];
      }
    }
  }
}

// fp32 W (K x N) -> bf16 W^T (N x K), LDS-tiled so both sides are coalesced.
__global__ __launch_bounds__(256) void wconv(const float* __restrict__ W,
                                             __bf16* __restrict__ WT, int K, int N){
  __shared__ float tile[32][33];
  const int tx = threadIdx.x & 31, ty = threadIdx.x >> 5;
  const int n0 = blockIdx.x * 32, k0 = blockIdx.y * 32;
  for (int i=ty; i<32; i+=8) tile[i][tx] = W[(size_t)(k0+i)*N + n0 + tx];
  __syncthreads();
  for (int i=ty; i<32; i+=8) WT[(size_t)(n0+i)*K + k0 + tx] = (__bf16)tile[tx][i];
}

// Row LayerNorm (E=512, 256 threads, 2 elems/thread), fp32 in -> bf16 out.
__global__ __launch_bounds__(256) void ln_row(const float* __restrict__ in,
                                              const float* __restrict__ g,
                                              const float* __restrict__ be,
                                              __bf16* __restrict__ out, int E){
  const int r = blockIdx.x, t = threadIdx.x;
  const float* x = in + (size_t)r*E;
  float x0 = x[t], x1 = x[t+256];
  __shared__ float red[256];
  red[t] = x0 + x1; __syncthreads();
  for (int o=128;o;o>>=1){ if (t<o) red[t] += red[t+o]; __syncthreads(); }
  const float mu = red[0] * (1.f/512.f); __syncthreads();
  float d0 = x0-mu, d1 = x1-mu;
  red[t] = d0*d0 + d1*d1; __syncthreads();
  for (int o=128;o;o>>=1){ if (t<o) red[t] += red[t+o]; __syncthreads(); }
  const float inv = rsqrtf(red[0]*(1.f/512.f) + 1e-5f);
  out[(size_t)r*E + t]     = (__bf16)(d0*inv*g[t]     + be[t]);
  out[(size_t)r*E + t+256] = (__bf16)(d1*inv*g[t+256] + be[t+256]);
}

// LN -> omega*pi -> write [cos||sin] bf16 (row length 1024).
__global__ __launch_bounds__(256) void ln_phase(const float* __restrict__ in,
                                                const float* __restrict__ g,
                                                const float* __restrict__ be,
                                                __bf16* __restrict__ out, int E){
  const int r = blockIdx.x, t = threadIdx.x;
  const float* x = in + (size_t)r*E;
  float x0 = x[t], x1 = x[t+256];
  __shared__ float red[256];
  red[t] = x0 + x1; __syncthreads();
  for (int o=128;o;o>>=1){ if (t<o) red[t] += red[t+o]; __syncthreads(); }
  const float mu = red[0] * (1.f/512.f); __syncthreads();
  float d0 = x0-mu, d1 = x1-mu;
  red[t] = d0*d0 + d1*d1; __syncthreads();
  for (int o=128;o;o>>=1){ if (t<o) red[t] += red[t+o]; __syncthreads(); }
  const float inv = rsqrtf(red[0]*(1.f/512.f) + 1e-5f);
  const float w0 = (d0*inv*g[t]     + be[t])     * PI_F;
  const float w1 = (d1*inv*g[t+256] + be[t+256]) * PI_F;
  __bf16* o0 = out + (size_t)r*1024;
  o0[t]           = (__bf16)__cosf(w0);  o0[512 + t]       = (__bf16)__sinf(w0);
  o0[t+256]       = (__bf16)__cosf(w1);  o0[512 + t + 256] = (__bf16)__sinf(w1);
}

// Holistic path: per-(b,e) online softmax over S with weighted sum of v.
__global__ __launch_bounds__(256) void holistic_reduce(const __bf16* __restrict__ s_,
                                                       const __bf16* __restrict__ v_,
                                                       float* __restrict__ qh_pre,
                                                       int S, int E){
  const int tx = threadIdx.x & 31, ty = threadIdx.x >> 5;
  const int e = blockIdx.x*32 + tx, b = blockIdx.y;
  float m = -3.4e38f, Z = 0.f, a = 0.f;
  for (int s = ty; s < S; s += 8){
    const size_t idx = ((size_t)b*S + s)*E + e;
    const float sv = (float)s_[idx], vv = (float)v_[idx];
    const float nm = fmaxf(m, sv);
    const float sc = __expf(m - nm), ev = __expf(sv - nm);
    Z = Z*sc + ev;  a = a*sc + ev*vv;  m = nm;
  }
  __shared__ float sm[8][32], sz[8][32], sa_[8][32];
  sm[ty][tx]=m; sz[ty][tx]=Z; sa_[ty][tx]=a;
  __syncthreads();
  if (ty == 0){
    for (int j=1;j<8;j++){
      const float mj=sm[j][tx], zj=sz[j][tx], aj=sa_[j][tx];
      const float nm=fmaxf(m,mj), s0=__expf(m-nm), s1=__expf(mj-nm);
      Z = Z*s0 + zj*s1;  a = a*s0 + aj*s1;  m = nm;
    }
    qh_pre[b*E + e] = a / Z;
  }
}

// q_a[b,e] = sum_s aw[b,s] * av[b,s,e]
__global__ __launch_bounds__(256) void colwsum(const float* __restrict__ aw,
                                               const __bf16* __restrict__ av,
                                               float* __restrict__ qa, int S, int E){
  const int tx = threadIdx.x & 31, ty = threadIdx.x >> 5;
  const int e = blockIdx.x*32 + tx, b = blockIdx.y;
  float a = 0.f;
  for (int s = ty; s < S; s += 8)
    a += aw[(size_t)b*S + s] * (float)av[((size_t)b*S + s)*E + e];
  __shared__ float sp[8][32];
  sp[ty][tx] = a; __syncthreads();
  if (ty == 0){
    for (int j=1;j<8;j++) a += sp[j][tx];
    qa[b*E + e] = a;
  }
}

// sc[r] = xa_row . as_w + as_b   (one wave per row)
__global__ __launch_bounds__(256) void matvec_sc(const __bf16* __restrict__ xa,
                                                 const float* __restrict__ w,
                                                 const float* __restrict__ b,
                                                 float* __restrict__ sc, int K){
  const int lane = threadIdx.x & 31, wv = threadIdx.x >> 5;
  const int row = blockIdx.x*8 + wv;
  const __bf16* xr = xa + (size_t)row*K;
  float acc = 0.f;
  for (int k = lane; k < K; k += 32) acc += (float)xr[k] * w[k];
  for (int o=16;o;o>>=1) acc += __shfl_down(acc, o, 32);
  if (lane == 0) sc[row] = acc + b[0];
}

// gates[r,0..2] = silu(xa_row @ gt_w + gt_b)
__global__ __launch_bounds__(256) void gates_mv(const __bf16* __restrict__ xa,
                                                const float* __restrict__ w,
                                                const float* __restrict__ b,
                                                float* __restrict__ gates, int K){
  const int lane = threadIdx.x & 31, wv = threadIdx.x >> 5;
  const int row = blockIdx.x*8 + wv;
  const __bf16* xr = xa + (size_t)row*K;
  float a0=0.f, a1=0.f, a2=0.f;
  for (int k = lane; k < K; k += 32){
    const float a = (float)xr[k];
    a0 += a*w[k*3+0]; a1 += a*w[k*3+1]; a2 += a*w[k*3+2];
  }
  for (int o=16;o;o>>=1){ a0 += __shfl_down(a0,o,32); a1 += __shfl_down(a1,o,32); a2 += __shfl_down(a2,o,32); }
  if (lane == 0){
    float g0=a0+b[0], g1=a1+b[1], g2=a2+b[2];
    gates[(size_t)row*3+0] = g0/(1.f+__expf(-g0));
    gates[(size_t)row*3+1] = g1/(1.f+__expf(-g1));
    gates[(size_t)row*3+2] = g2/(1.f+__expf(-g2));
  }
}

// scalar softmax over S per batch (4 blocks)
__global__ __launch_bounds__(256) void softmax_seq(const float* __restrict__ sc,
                                                   float* __restrict__ aw, int S){
  const int b = blockIdx.x, t = threadIdx.x;
  __shared__ float red[256];
  float m = -3.4e38f;
  for (int s=t; s<S; s+=256) m = fmaxf(m, sc[(size_t)b*S+s]);
  red[t]=m; __syncthreads();
  for (int o=128;o;o>>=1){ if (t<o) red[t]=fmaxf(red[t],red[t+o]); __syncthreads(); }
  m = red[0]; __syncthreads();
  float z = 0.f;
  for (int s=t; s<S; s+=256) z += __expf(sc[(size_t)b*S+s]-m);
  red[t]=z; __syncthreads();
  for (int o=128;o;o>>=1){ if (t<o) red[t]+=red[t+o]; __syncthreads(); }
  const float inv = 1.f/red[0];
  for (int s=t; s<S; s+=256) aw[(size_t)b*S+s] = __expf(sc[(size_t)b*S+s]-m)*inv;
}

// tiny GEMM for q_h = qh_pre @ hc_w + hc_b  (4x512x512)
__global__ void small_mm(const float* __restrict__ A, const float* __restrict__ W,
                         const float* __restrict__ bias, float* __restrict__ out,
                         int Bv, int K, int N){
  const int id = blockIdx.x*blockDim.x + threadIdx.x;
  if (id >= Bv*N) return;
  const int b = id / N, n = id % N;
  float acc = bias[n];
  for (int k=0;k<K;k++) acc += A[b*K+k] * W[(size_t)k*N+n];
  out[id] = acc;
}

// 3-phase chunked cumsum of sigmoid-gated values, fp32 accumulation.
__global__ __launch_bounds__(64) void scanA(const __bf16* __restrict__ sg,
                                            const __bf16* __restrict__ vs,
                                            float* __restrict__ ctot,
                                            int S, int E, int CL, int NC){
  const int e = blockIdx.x*64 + threadIdx.x, c = blockIdx.y, b = blockIdx.z;
  const size_t base = ((size_t)b*S + (size_t)c*CL)*E + e;
  float s = 0.f;
  for (int i=0;i<CL;i++){ const size_t idx = base + (size_t)i*E; s += (float)sg[idx]*(float)vs[idx]; }
  ctot[((size_t)b*E + e)*NC + c] = s;
}
__global__ void scanB(float* __restrict__ ctot, int NC, int total){
  const int id = blockIdx.x*blockDim.x + threadIdx.x;
  if (id >= total) return;
  float* p = ctot + (size_t)id*NC;
  float run = 0.f;
  for (int c=0;c<NC;c++){ const float v = p[c]; p[c] = run; run += v; }
}
__global__ __launch_bounds__(64) void scanC(const __bf16* __restrict__ sg,
                                            const __bf16* __restrict__ vs,
                                            const float* __restrict__ ctot,
                                            __bf16* __restrict__ cum,
                                            int S, int E, int CL, int NC, float invS){
  const int e = blockIdx.x*64 + threadIdx.x, c = blockIdx.y, b = blockIdx.z;
  const size_t base = ((size_t)b*S + (size_t)c*CL)*E + e;
  float run = ctot[((size_t)b*E + e)*NC + c];
  for (int i=0;i<CL;i++){
    const size_t idx = base + (size_t)i*E;
    run += (float)sg[idx]*(float)vs[idx];
    cum[idx] = (__bf16)(run * invS);
  }
}

// h = g0*qh[b] + g1*qa[b] + g2*qs[r]; then LayerNorm(n_g,n_b) -> bf16
__global__ __launch_bounds__(256) void mix_ln(const float* __restrict__ gates,
                                              const float* __restrict__ qh,
                                              const float* __restrict__ qa,
                                              const float* __restrict__ qs,
                                              const float* __restrict__ g,
                                              const float* __restrict__ be,
                                              __bf16* __restrict__ out, int S, int E){
  const int r = blockIdx.x, t = threadIdx.x, b = r / S;
  const float g0 = gates[(size_t)r*3+0], g1 = gates[(size_t)r*3+1], g2 = gates[(size_t)r*3+2];
  const int e0 = t, e1 = t + 256;
  float h0 = g0*qh[b*E+e0] + g1*qa[b*E+e0] + g2*qs[(size_t)r*E+e0];
  float h1 = g0*qh[b*E+e1] + g1*qa[b*E+e1] + g2*qs[(size_t)r*E+e1];
  __shared__ float red[256];
  red[t] = h0 + h1; __syncthreads();
  for (int o=128;o;o>>=1){ if (t<o) red[t] += red[t+o]; __syncthreads(); }
  const float mu = red[0]*(1.f/512.f); __syncthreads();
  const float d0 = h0-mu, d1 = h1-mu;
  red[t] = d0*d0 + d1*d1; __syncthreads();
  for (int o=128;o;o>>=1){ if (t<o) red[t] += red[t+o]; __syncthreads(); }
  const float inv = rsqrtf(red[0]*(1.f/512.f) + 1e-5f);
  out[(size_t)r*E + e0] = (__bf16)(d0*inv*g[e0] + be[e0]);
  out[(size_t)r*E + e1] = (__bf16)(d1*inv*g[e1] + be[e1]);
}

// ---------------------------------------------------------------------------
extern "C" void kernel_launch(void* const* d_in, const int* in_sizes, int n_in,
                              void* d_out, int out_size, void* d_ws, size_t ws_size,
                              hipStream_t stream) {
  constexpr int Bv = 4, S = 8192, E = 512, F = 2048, M = Bv*S;
  constexpr int NC = 32, CL = S/NC;

  const float* x     = (const float*)d_in[0];
  const float* ad_w  = (const float*)d_in[1];  const float* ad_b = (const float*)d_in[2];
  const float* ad_g  = (const float*)d_in[3];  const float* ad_be= (const float*)d_in[4];
  const float* hs_w  = (const float*)d_in[5];  const float* hs_b = (const float*)d_in[6];
  const float* hv_w  = (const float*)d_in[7];  const float* hv_b = (const float*)d_in[8];
  const float* hc_w  = (const float*)d_in[9];  const float* hc_b = (const float*)d_in[10];
  const float* as_w  = (const float*)d_in[11]; const float* as_b = (const float*)d_in[12];
  const float* av_w  = (const float*)d_in[13]; const float* av_b = (const float*)d_in[14];
  const float* ss_w  = (const float*)d_in[15]; const float* ss_b = (const float*)d_in[16];
  const float* sv_w  = (const float*)d_in[17]; const float* sv_b = (const float*)d_in[18];
  const float* sa_w  = (const float*)d_in[19]; const float* sa_b = (const float*)d_in[20];
  const float* sln_g = (const float*)d_in[21]; const float* sln_b= (const float*)d_in[22];
  const float* so_w  = (const float*)d_in[23]; const float* so_b = (const float*)d_in[24];
  const float* gt_w  = (const float*)d_in[25]; const float* gt_b = (const float*)d_in[26];
  const float* f1_w  = (const float*)d_in[27]; const float* f1_b = (const float*)d_in[28];
  const float* f2_w  = (const float*)d_in[29]; const float* f2_b = (const float*)d_in[30];
  const float* n_g   = (const float*)d_in[31]; const float* n_b  = (const float*)d_in[32];
  float* out = (float*)d_out;

  // -------- workspace layout (bump allocator, ~426 MB, overlays noted) -----
  char* ws = (char*)d_ws;
  size_t off = 0;
  auto alloc = [&](size_t bytes)->char*{ char* p = ws + off; off = (off + bytes + 255) & ~(size_t)255; return p; };

  __bf16* wt_ad = (__bf16*)alloc((size_t)E*E*2);
  __bf16* wt_hs = (__bf16*)alloc((size_t)E*E*2);
  __bf16* wt_hv = (__bf16*)alloc((size_t)E*E*2);
  __bf16* wt_av = (__bf16*)alloc((size_t)E*E*2);
  __bf16* wt_ss = (__bf16*)alloc((size_t)E*E*2);
  __bf16* wt_sv = (__bf16*)alloc((size_t)E*E*2);
  __bf16* wt_sa = (__bf16*)alloc((size_t)E*E*2);
  __bf16* wt_so = (__bf16*)alloc((size_t)2*E*E*2);
  __bf16* wt_f1 = (__bf16*)alloc((size_t)E*F*2);
  __bf16* wt_f2 = (__bf16*)alloc((size_t)F*E*2);

  __bf16* xa     = (__bf16*)alloc((size_t)M*E*2);    // adapter output, bf16
  __bf16* bufS   = (__bf16*)alloc((size_t)M*E*2);    // hs scores
  __bf16* bufV   = (__bf16*)alloc((size_t)M*E*2);    // hv values (contiguous after bufS)
  __bf16* bufAV  = (__bf16*)alloc((size_t)M*E*2);    // av values
  __bf16* bufSG  = (__bf16*)alloc((size_t)M*E*2);    // sigmoid gate
  __bf16* bufVS  = (__bf16*)alloc((size_t)M*E*2);    // sv values (contiguous after bufSG)
  __bf16* bufCUM = (__bf16*)alloc((size_t)M*E*2);    // cumsum/S
  float*  t1     = (float*)alloc((size_t)M*E*4);     // fp32 scratch (ad pre-LN, later sa out)
  __bf16* f1out  = (__bf16*)alloc((size_t)M*F*2);    // FFN hidden
  float*  qh_pre = (float*)alloc((size_t)Bv*E*4);
  float*  qh     = (float*)alloc((size_t)Bv*E*4);
  float*  qa     = (float*)alloc((size_t)Bv*E*4);
  float*  sc     = (float*)alloc((size_t)M*4);
  float*  aw     = (float*)alloc((size_t)M*4);
  float*  gates  = (float*)alloc((size_t)M*3*4);
  float*  ctot   = (float*)alloc((size_t)Bv*E*NC*4);
  // overlays of dead buffers:
  __bf16* cs  = bufS;            // cos||sin (M x 1024 bf16 = 64MB over bufS+bufV)
  float*  qs  = (float*)bufSG;   // q_s fp32 (64MB over bufSG+bufVS)
  __bf16* hln = bufAV;           // LN(h) bf16 (32MB over bufAV)

  // -------- weight conversion fp32 (KxN) -> bf16 transposed (NxK) ----------
  wconv<<<dim3(E/32,   E/32),   256, 0, stream>>>(ad_w, wt_ad, E,   E);
  wconv<<<dim3(E/32,   E/32),   256, 0, stream>>>(hs_w, wt_hs, E,   E);
  wconv<<<dim3(E/32,   E/32),   256, 0, stream>>>(hv_w, wt_hv, E,   E);
  wconv<<<dim3(E/32,   E/32),   256, 0, stream>>>(av_w, wt_av, E,   E);
  wconv<<<dim3(E/32,   E/32),   256, 0, stream>>>(ss_w, wt_ss, E,   E);
  wconv<<<dim3(E/32,   E/32),   256, 0, stream>>>(sv_w, wt_sv, E,   E);
  wconv<<<dim3(E/32,   E/32),   256, 0, stream>>>(sa_w, wt_sa, E,   E);
  wconv<<<dim3(E/32, 2*E/32),   256, 0, stream>>>(so_w, wt_so, 2*E, E);
  wconv<<<dim3(F/32,   E/32),   256, 0, stream>>>(f1_w, wt_f1, E,   F);
  wconv<<<dim3(E/32,   F/32),   256, 0, stream>>>(f2_w, wt_f2, F,   E);

  const dim3 blk(256);
  const dim3 gE(M/128, E/128), gF(M/128, F/128);

  // input adapter: GELU(x@ad_w+b) -> LN -> xa (bf16)
  gemm_wmma<float,  EPI_GELU_F32><<<gE, blk, 0, stream>>>(x,  wt_ad, ad_b, nullptr, t1, M, E, E);
  ln_row<<<M, 256, 0, stream>>>(t1, ad_g, ad_be, xa, E);

  // holistic extractor
  gemm_wmma<__bf16, EPI_BF16><<<gE, blk, 0, stream>>>(xa, wt_hs, hs_b, nullptr, bufS, M, E, E);
  gemm_wmma<__bf16, EPI_BF16><<<gE, blk, 0, stream>>>(xa, wt_hv, hv_b, nullptr, bufV, M, E, E);
  holistic_reduce<<<dim3(E/32, Bv), 256, 0, stream>>>(bufS, bufV, qh_pre, S, E);
  small_mm<<<(Bv*E + 255)/256, 256, 0, stream>>>(qh_pre, hc_w, hc_b, qh, Bv, E, E);

  // associative extractor
  matvec_sc<<<M/8, 256, 0, stream>>>(xa, as_w, as_b, sc, E);
  softmax_seq<<<Bv, 256, 0, stream>>>(sc, aw, S);
  gemm_wmma<__bf16, EPI_BF16><<<gE, blk, 0, stream>>>(xa, wt_av, av_b, nullptr, bufAV, M, E, E);
  colwsum<<<dim3(E/32, Bv), 256, 0, stream>>>(aw, bufAV, qa, S, E);

  // sequential extractor
  gemm_wmma<__bf16, EPI_SIG_BF16><<<gE, blk, 0, stream>>>(xa, wt_ss, ss_b, nullptr, bufSG, M, E, E);
  gemm_wmma<__bf16, EPI_BF16    ><<<gE, blk, 0, stream>>>(xa, wt_sv, sv_b, nullptr, bufVS, M, E, E);
  scanA<<<dim3(E/64, NC, Bv), 64, 0, stream>>>(bufSG, bufVS, ctot, S, E, CL, NC);
  scanB<<<(Bv*E + 255)/256, 256, 0, stream>>>(ctot, NC, Bv*E);
  scanC<<<dim3(E/64, NC, Bv), 64, 0, stream>>>(bufSG, bufVS, ctot, bufCUM, S, E, CL, NC, 1.f/(float)S);
  gemm_wmma<__bf16, EPI_F32><<<gE, blk, 0, stream>>>(bufCUM, wt_sa, sa_b, nullptr, t1, M, E, E);
  ln_phase<<<M, 256, 0, stream>>>(t1, sln_g, sln_b, cs, E);
  gemm_wmma<__bf16, EPI_F32><<<gE, blk, 0, stream>>>(cs, wt_so, so_b, nullptr, qs, M, E, 2*E);

  // gate + mixture + pre-FFN LN
  gates_mv<<<M/8, 256, 0, stream>>>(xa, gt_w, gt_b, gates, E);
  mix_ln<<<M, 256, 0, stream>>>(gates, qh, qa, qs, n_g, n_b, hln, S, E);

  // FFN with fused bias + residual
  gemm_wmma<__bf16, EPI_GELU_BF16><<<gF, blk, 0, stream>>>(hln,   wt_f1, f1_b, nullptr, f1out, M, F, E);
  gemm_wmma<__bf16, EPI_RESID_F32><<<gE, blk, 0, stream>>>(f1out, wt_f2, f2_b, x,       out,   M, E, F);
}
